// NDLS_6193342841620
// MI455X (gfx1250) — matrix operand order repaired and checked by reading, not compiled, not run on hardware
//
#include <hip/hip_runtime.h>
#include <hip/hip_bf16.h>
#include <cstdint>

// ---------------------------------------------------------------------------
// Problem constants (match reference)
// ---------------------------------------------------------------------------
#define K1     16
#define EPS1   0.03f
#define ALPHA  0.15f
#define DIMF   64
#define HID    256
#define NCLS   40
#define NCLSP  48      // padded to 3x16 WMMA tiles

typedef __attribute__((ext_vector_type(16))) _Float16 v16h;
typedef __attribute__((ext_vector_type(8)))  _Float16 v8h;
typedef __attribute__((ext_vector_type(8)))  float    v8f;
typedef __attribute__((ext_vector_type(4)))  int      v4i;

#if defined(__has_builtin)
#if __has_builtin(__builtin_amdgcn_global_load_async_to_lds_b128)
#define HAVE_ASYNC_LDS 1
#endif
#endif

// ---------------------------------------------------------------------------
// WMMA fragment helpers (CDNA5 ISA 7.12.2, wave32)
// A (16x32 f16, row-major M x K src): lane = M (mod 16); per lane the 16
//   slots are two contiguous runs: k0+half*8..+8 and k0+16+half*8..+8.
// B (32x16 f16): lane = N (mod 16); same two k runs per lane as A:
//   k0+half*8..+8 and k0+16+half*8..+8. With column-major [N][K] weight storage this
//   is two contiguous 8xf16 runs per lane.
// C (16x16 f32): lane = N (mod 16); VGPR v -> M = v + 8*(lane>=16).
// ---------------------------------------------------------------------------
__device__ __forceinline__ v16h cat8(v8h lo, v8h hi) {
    return __builtin_shufflevector(lo, hi, 0, 1, 2, 3, 4, 5, 6, 7,
                                   8, 9, 10, 11, 12, 13, 14, 15);
}

__device__ __forceinline__ v8f wmma_f16(v16h a, v16h b, v8f c) {
    return __builtin_amdgcn_wmma_f32_16x16x32_f16(
        /*neg_a=*/false, a, /*neg_b=*/false, b,
        /*c_mod=*/(short)0, c, /*reuse_a=*/false, /*reuse_b=*/false);
}

__device__ __forceinline__ void lds_fence() {
    asm volatile("s_wait_dscnt 0" ::: "memory");
}

__device__ __forceinline__ void wait_async0() {
#if defined(__has_builtin)
#if __has_builtin(__builtin_amdgcn_s_wait_asynccnt)
    __builtin_amdgcn_s_wait_asynccnt(0);
#else
    asm volatile("s_wait_asynccnt 0" ::: "memory");
#endif
#else
    asm volatile("s_wait_asynccnt 0" ::: "memory");
#endif
}

// A-fragment load from row-major [M][ld] f16 (ld, k0, half*8 all 8-f16 mult)
__device__ __forceinline__ v16h load_a(const _Float16* __restrict__ p,
                                       int ld, int k0, int lane) {
    const int half = lane >> 4, m = lane & 15;
    const _Float16* base = p + (size_t)m * ld + k0 + half * 8;
    v8h lo = *(const v8h*)(base);
    v8h hi = *(const v8h*)(base + 16);
    return cat8(lo, hi);
}

// B-fragment load from column-major [N][Krows] f16 weight storage
__device__ __forceinline__ v16h load_bt(const _Float16* __restrict__ p,
                                        int k0, int n0, int Krows, int lane) {
    const int half = lane >> 4, n = lane & 15;
    const _Float16* base = p + (size_t)(n0 + n) * Krows + k0 + half * 8;
    v8h lo = *(const v8h*)(base);
    v8h hi = *(const v8h*)(base + 16);
    return cat8(lo, hi);
}

// ---------------------------------------------------------------------------
// Graph-phase kernels
// ---------------------------------------------------------------------------

__global__ void k_deg(const int* __restrict__ rows, float* __restrict__ deg, int e) {
    int i = blockIdx.x * blockDim.x + threadIdx.x;
    if (i < e) atomicAdd(&deg[rows[i]], 1.0f);
}

// L1-normalize rows of h -> feat (and x = F[0]); accumulate f_inf partials
__global__ void k_feat(const float* __restrict__ h, const float* __restrict__ deg,
                       float* __restrict__ feat, float* __restrict__ x,
                       float* __restrict__ f_inf, int n) {
    __shared__ float sfin[DIMF];
    int tid = threadIdx.x;
    if (tid < DIMF) sfin[tid] = 0.0f;
    __syncthreads();

    int lane = tid & 31;
    int row  = blockIdx.x * 8 + (tid >> 5);
    if (row < n) {
        size_t base = (size_t)row * DIMF;
        float a0 = h[base + lane];
        float a1 = h[base + 32 + lane];
        float s = fabsf(a0) + fabsf(a1);
#pragma unroll
        for (int m = 1; m < 32; m <<= 1) s += __shfl_xor(s, m, 32);
        float inv = 1.0f / fmaxf(s, 1e-12f);
        float f0 = a0 * inv, f1 = a1 * inv;
        feat[base + lane] = f0;        feat[base + 32 + lane] = f1;
        x[base + lane]    = f0;        x[base + 32 + lane]    = f1;
        float sld = deg[row] + 1.0f;
        atomicAdd(&sfin[lane],      sld * f0);
        atomicAdd(&sfin[32 + lane], sld * f1);
    }
    __syncthreads();
    if (tid < DIMF) atomicAdd(&f_inf[tid], sfin[tid]);
}

__global__ void k_scale_finf(float* __restrict__ f_inf, float scale) {
    if (threadIdx.x < DIMF) f_inf[threadIdx.x] *= scale;
}

// Streaming hop check: on first ||F[k]-f_inf|| < eps (or forced last step)
// capture Chold = C (== C[k-1]); otherwise C += F[k].
__global__ void k_check(const float* __restrict__ x, const float* __restrict__ f_inf,
                        float* __restrict__ C, float* __restrict__ Chold,
                        int* __restrict__ hops, int k, int last, int n) {
    int tid  = threadIdx.x;
    int lane = tid & 31;
    int row  = blockIdx.x * 8 + (tid >> 5);
    if (row >= n) return;
    if (hops[row] >= 0) return;

    size_t base = (size_t)row * DIMF;
    float x0 = x[base + lane], x1 = x[base + 32 + lane];
    float d0 = x0 - f_inf[lane], d1 = x1 - f_inf[32 + lane];
    float s = d0 * d0 + d1 * d1;
#pragma unroll
    for (int m = 1; m < 32; m <<= 1) s += __shfl_xor(s, m, 32);

    if (s < (EPS1 * EPS1) || last) {
        Chold[base + lane]      = C[base + lane];
        Chold[base + 32 + lane] = C[base + 32 + lane];
        if (lane == 0) hops[row] = k;
    } else {
        C[base + lane]      += x0;
        C[base + 32 + lane] += x1;
    }
}

// edge scatter: ax[rows[e]] += x[cols[e]]  (16 threads/edge, float4 each)
__global__ void k_scatter(const int* __restrict__ rows, const int* __restrict__ cols,
                          const float* __restrict__ x, float* __restrict__ ax,
                          int e) {
    int t = blockIdx.x * blockDim.x + threadIdx.x;
    int ei = t >> 4;
    int g  = t & 15;
    if (ei >= e) return;
    if (g == 0) {  // stream-ahead prefetch of the edge lists (global_prefetch_b8)
        __builtin_prefetch(rows + ei + 4096, 0, 0);
        __builtin_prefetch(cols + ei + 4096, 0, 0);
    }
    int r = rows[ei], c = cols[ei];
    const float4 v = *(const float4*)(x + (size_t)c * DIMF + g * 4);
    float* dst = ax + (size_t)r * DIMF + g * 4;
    atomicAdd(dst + 0, v.x);
    atomicAdd(dst + 1, v.y);
    atomicAdd(dst + 2, v.z);
    atomicAdd(dst + 3, v.w);
}

// x = (x + ax) / (deg + 1)
__global__ void k_combine(float* __restrict__ x, const float* __restrict__ ax,
                          const float* __restrict__ deg, int n) {
    size_t i = (size_t)blockIdx.x * blockDim.x + threadIdx.x;
    if (i >= (size_t)n * DIMF) return;
    float dinv = 1.0f / (deg[i / DIMF] + 1.0f);
    x[i] = (x[i] + ax[i]) * dinv;
}

// final NDLS feature -> f16 for WMMA MLP
__global__ void k_final(const float* __restrict__ feat, const float* __restrict__ Chold,
                        const int* __restrict__ hops, _Float16* __restrict__ feah,
                        int n) {
    size_t i = (size_t)blockIdx.x * blockDim.x + threadIdx.x;
    if (i >= (size_t)n * DIMF) return;
    int hp = hops[i / DIMF];
    float f = feat[i];
    float val;
    if (hp == 0) {
        val = f;
    } else {
        float hf = fmaxf((float)hp, 1.0f);
        val = (1.0f - ALPHA) * Chold[i] / hf + ALPHA * f;
    }
    feah[i] = (_Float16)val;
}

// weights -> f16 TRANSPOSED (column-major [N][K]) so B-fragment loads are
// contiguous 128-bit LDS reads. W3 padded to 48 output cols with zeros.
__global__ void k_wconv(const float* __restrict__ W1, const float* __restrict__ W2,
                        const float* __restrict__ W3,
                        _Float16* __restrict__ W1t, _Float16* __restrict__ W2t,
                        _Float16* __restrict__ W3t) {
    int i = blockIdx.x * blockDim.x + threadIdx.x;
    if (i < HID * DIMF) {                 // W1t[n][k], n<256, k<64
        int nn = i / DIMF, kk = i % DIMF;
        W1t[i] = (_Float16)W1[(size_t)kk * HID + nn];
    }
    if (i < HID * HID) {                  // W2t[n][k], n<256, k<256
        int nn = i / HID, kk = i % HID;
        W2t[i] = (_Float16)W2[(size_t)kk * HID + nn];
    }
    if (i < NCLSP * HID) {                // W3t[n][k], n<48, k<256
        int nn = i / HID, kk = i % HID;
        W3t[i] = (nn < NCLS) ? (_Float16)W3[(size_t)kk * NCLS + nn] : (_Float16)0.0f;
    }
}

// ---------------------------------------------------------------------------
// MLP kernel: all weights LDS-resident (CDNA5: 320KB/WGP), one wave per
// 16-row node tile, WMMA f32_16x16x32_f16 chains, log-softmax via shuffles.
// blockDim = 128 (4 waves).
// ---------------------------------------------------------------------------
#define MLP_WAVES 4
#define WF16   (HID * DIMF + HID * HID + NCLSP * HID)   // f16 weight elements
#define WBYTES (WF16 * 2)

__global__ void k_mlp(const _Float16* __restrict__ feah,
                      const _Float16* __restrict__ W1t, const float* __restrict__ b1,
                      const _Float16* __restrict__ W2t, const float* __restrict__ b2,
                      const _Float16* __restrict__ W3t, const float* __restrict__ b3,
                      float* __restrict__ out, int n, int ntiles) {
    extern __shared__ char smem_raw[];
    _Float16* sW1 = (_Float16*)smem_raw;                 // 256*64  (col-major)
    _Float16* sW2 = sW1 + HID * DIMF;                    // 256*256 (col-major)
    _Float16* sW3 = sW2 + HID * HID;                     // 48*256  (col-major)
    float*    sb1 = (float*)(sW3 + NCLSP * HID);         // 256
    float*    sb2 = sb1 + HID;                           // 256
    float*    sb3 = sb2 + HID;                           // 48
    _Float16* stage = (_Float16*)(sb3 + NCLSP);          // waves * 2 * 16*256

    const int tid = threadIdx.x, nthr = blockDim.x;

    // ---- weight preload: gfx1250 async global->LDS path when available ----
#if defined(HAVE_ASYNC_LDS)
    {
        // builtin signature: (int4 AS1*, int4 AS3*, imm offset, imm cpol)
        typedef __attribute__((address_space(1))) v4i g_v4i;
        typedef __attribute__((address_space(3))) v4i l_v4i;
        const char* gbase = (const char*)W1t;                 // W1t/W2t/W3t contiguous
        const uint32_t lbase = (uint32_t)(uintptr_t)(char*)sW1;  // flat->LDS: low 32 bits
        for (int off = tid * 16; off < WBYTES; off += nthr * 16) {
            g_v4i* gp = (g_v4i*)(uintptr_t)(gbase + off);
            l_v4i* lp = (l_v4i*)(lbase + off);
            __builtin_amdgcn_global_load_async_to_lds_b128(gp, lp, 0, 0);
        }
        wait_async0();
    }
#else
    {
        const v8h* gsrc = (const v8h*)W1t;
        v8h*       ldst = (v8h*)sW1;
        for (int i = tid; i < WF16 / 8; i += nthr) ldst[i] = gsrc[i];
    }
#endif
    for (int i = tid; i < HID; i += nthr) { sb1[i] = b1[i]; sb2[i] = b2[i]; }
    for (int i = tid; i < NCLSP; i += nthr) sb3[i] = (i < NCLS) ? b3[i] : 0.0f;
    __syncthreads();

    const int wave = tid >> 5, lane = tid & 31;
    const int half = lane >> 4, lm = lane & 15;
    _Float16* x1s = stage + (size_t)wave * (2 * 16 * HID);
    _Float16* x2s = x1s + 16 * HID;

    const int gw = blockIdx.x * MLP_WAVES + wave;
    const int nw = gridDim.x * MLP_WAVES;

    for (int t = gw; t < ntiles; t += nw) {
        const int row0 = t * 16;

        // ----- layer 1: X(16x64) @ W1(64x256) -----
        v16h a0, a1;
        {
            int row = row0 + lm; if (row >= n) row = n - 1;
            const _Float16* p = feah + (size_t)row * DIMF + half * 8;
            a0 = cat8(*(const v8h*)(p),      *(const v8h*)(p + 16));
            a1 = cat8(*(const v8h*)(p + 32), *(const v8h*)(p + 48));
        }
#pragma unroll
        for (int nt = 0; nt < 16; ++nt) {
            float bias = sb1[nt * 16 + lm];
            v8f acc;
#pragma unroll
            for (int i = 0; i < 8; ++i) acc[i] = bias;
            acc = wmma_f16(a0, load_bt(sW1, 0,  nt * 16, DIMF, lane), acc);
            acc = wmma_f16(a1, load_bt(sW1, 32, nt * 16, DIMF, lane), acc);
#pragma unroll
            for (int v = 0; v < 8; ++v)
                x1s[(size_t)(v + half * 8) * HID + nt * 16 + lm] =
                    (_Float16)fmaxf(acc[v], 0.0f);
        }
        lds_fence();

        // ----- layer 2: x1(16x256) @ W2(256x256) -----
        v16h a2[8];
#pragma unroll
        for (int kc = 0; kc < 8; ++kc) a2[kc] = load_a(x1s, HID, kc * 32, lane);
#pragma unroll 2
        for (int nt = 0; nt < 16; ++nt) {
            float bias = sb2[nt * 16 + lm];
            v8f acc;
#pragma unroll
            for (int i = 0; i < 8; ++i) acc[i] = bias;
#pragma unroll
            for (int kc = 0; kc < 8; ++kc)
                acc = wmma_f16(a2[kc], load_bt(sW2, kc * 32, nt * 16, HID, lane), acc);
#pragma unroll
            for (int v = 0; v < 8; ++v)
                x2s[(size_t)(v + half * 8) * HID + nt * 16 + lm] =
                    (_Float16)fmaxf(acc[v], 0.0f);
        }
        lds_fence();

        // ----- layer 3: x2(16x256) @ W3(256x48) + log_softmax(40) -----
        v16h a3[8];
#pragma unroll
        for (int kc = 0; kc < 8; ++kc) a3[kc] = load_a(x2s, HID, kc * 32, lane);
        float r0[8], r1[8], r2[8];
#pragma unroll
        for (int nt = 0; nt < 3; ++nt) {
            float bias = sb3[nt * 16 + lm];
            v8f acc;
#pragma unroll
            for (int i = 0; i < 8; ++i) acc[i] = bias;
#pragma unroll
            for (int kc = 0; kc < 8; ++kc)
                acc = wmma_f16(a3[kc], load_bt(sW3, kc * 32, nt * 16, HID, lane), acc);
#pragma unroll
            for (int v = 0; v < 8; ++v) {
                if (nt == 0) r0[v] = acc[v];
                else if (nt == 1) r1[v] = acc[v];
                else r2[v] = acc[v];
            }
        }
        const bool v2ok = (lm < 8);  // class 32+lm valid only if < 40
#pragma unroll
        for (int v = 0; v < 8; ++v) {
            float e2 = v2ok ? r2[v] : -3.4e38f;
            float mx = fmaxf(fmaxf(r0[v], r1[v]), e2);
#pragma unroll
            for (int m = 1; m < 16; m <<= 1) mx = fmaxf(mx, __shfl_xor(mx, m, 32));
            float se = __expf(r0[v] - mx) + __expf(r1[v] - mx) +
                       (v2ok ? __expf(r2[v] - mx) : 0.0f);
#pragma unroll
            for (int m = 1; m < 16; m <<= 1) se += __shfl_xor(se, m, 32);
            float lse = mx + __logf(se);
            int row = row0 + v + half * 8;
            if (row < n) {
                float* o = out + (size_t)row * NCLS;
                o[lm]      = r0[v] - lse;
                o[16 + lm] = r1[v] - lse;
                if (v2ok) o[32 + lm] = r2[v] - lse;
            }
        }
    }
}

// ---------------------------------------------------------------------------
// Host launcher
// ---------------------------------------------------------------------------
extern "C" void kernel_launch(void* const* d_in, const int* in_sizes, int n_in,
                              void* d_out, int out_size, void* d_ws, size_t ws_size,
                              hipStream_t stream) {
    const float* h    = (const float*)d_in[0];
    const int*   rows = (const int*)  d_in[1];
    const int*   cols = (const int*)  d_in[2];
    const float* W1   = (const float*)d_in[3];
    const float* b1   = (const float*)d_in[4];
    const float* W2   = (const float*)d_in[5];
    const float* b2   = (const float*)d_in[6];
    const float* W3   = (const float*)d_in[7];
    const float* b3   = (const float*)d_in[8];
    float* out = (float*)d_out;

    const int N = in_sizes[0] / DIMF;
    const int E = in_sizes[1];
    const size_t ND = (size_t)N * DIMF;

    // workspace layout (all offsets stay 16B aligned)
    float*    deg   = (float*)d_ws;            // N
    float*    feat  = deg   + N;               // N*64
    float*    x     = feat  + ND;              // N*64
    float*    ax    = x     + ND;              // N*64
    float*    C     = ax    + ND;              // N*64
    float*    Chold = C     + ND;              // N*64
    float*    f_inf = Chold + ND;              // 64
    int*      hops  = (int*)(f_inf + DIMF);    // N
    _Float16* W1t   = (_Float16*)(hops + N);   // 256*64  (col-major)
    _Float16* W2t   = W1t + HID * DIMF;        // 256*256 (col-major)
    _Float16* W3t   = W2t + HID * HID;         // 48*256  (col-major)
    _Float16* feah  = W3t + NCLSP * HID;       // N*64

    (void)hipMemsetAsync(deg,   0,    (size_t)N * sizeof(float), stream);
    (void)hipMemsetAsync(f_inf, 0,    DIMF * sizeof(float), stream);
    (void)hipMemsetAsync(C,     0,    ND * sizeof(float), stream);
    (void)hipMemsetAsync(hops,  0xFF, (size_t)N * sizeof(int), stream);  // -1

    k_deg<<<(E + 255) / 256, 256, 0, stream>>>(rows, deg, E);
    k_feat<<<(N + 7) / 8, 256, 0, stream>>>(h, deg, feat, x, f_inf, N);
    k_scale_finf<<<1, 64, 0, stream>>>(f_inf, 1.0f / (2.0f * (float)E + (float)N));
    k_wconv<<<(HID * HID + 255) / 256, 256, 0, stream>>>(W1, W2, W3, W1t, W2t, W3t);

    for (int k = 0; k < K1; ++k) {
        k_check<<<(N + 7) / 8, 256, 0, stream>>>(x, f_inf, C, Chold, hops, k,
                                                 (k == K1 - 1) ? 1 : 0, N);
        if (k < K1 - 1) {
            (void)hipMemsetAsync(ax, 0, ND * sizeof(float), stream);
            size_t sthreads = (size_t)E * 16;
            k_scatter<<<(int)((sthreads + 255) / 256), 256, 0, stream>>>(rows, cols, x, ax, E);
            k_combine<<<(int)((ND + 255) / 256), 256, 0, stream>>>(x, ax, deg, N);
        }
    }

    k_final<<<(int)((ND + 255) / 256), 256, 0, stream>>>(feat, Chold, hops, feah, N);

    const int ntiles = (N + 15) / 16;
    const size_t smem =
        (size_t)WF16 * sizeof(_Float16) +
        (size_t)(HID + HID + NCLSP) * sizeof(float) +
        (size_t)MLP_WAVES * 2 * 16 * HID * sizeof(_Float16);  // ~256 KB (CDNA5 WGP LDS)
    k_mlp<<<128, MLP_WAVES * 32, smem, stream>>>(feah, W1t, b1, W2t, b2, W3t, b3,
                                                 out, N, ntiles);
}
